// Codebook_11278584119805
// MI455X (gfx1250) — matrix operand order repaired and checked by reading, not compiled, not run on hardware
//
#include <hip/hip_runtime.h>
#include <hip/hip_bf16.h>
#include <stdint.h>

typedef __attribute__((ext_vector_type(16))) __bf16   v16bf;
typedef __attribute__((ext_vector_type(8)))  __bf16   v8bf;
typedef __attribute__((ext_vector_type(16))) uint16_t v16us;
typedef __attribute__((ext_vector_type(8)))  float    v8f;

#define D_MODEL        512
#define KCODES         4096
#define ROWS_PER_WG    64
#define THREADS_PER_WG 128            // 4 waves (wave32)
#define B_ROW_BYTES    1040           // 512 halfs + 16B pad -> rotates LDS banks
#define B_BUF_BYTES    (64 * B_ROW_BYTES)   // 32 hi-rows + 32 lo-rows = 66560 B

__device__ __forceinline__ uint16_t f32_to_bf16_rne(float x) {
    uint32_t u = __float_as_uint(x);
    uint32_t r = u + 0x7FFFu + ((u >> 16) & 1u);
    return (uint16_t)(r >> 16);
}
__device__ __forceinline__ float bf16_bits_to_f32(uint16_t h) {
    return __uint_as_float(((uint32_t)h) << 16);
}

// Low 32 bits of a generic LDS pointer are the LDS byte offset (ISA flat->LDS rule).
__device__ __forceinline__ uint32_t lds_offset(const void* p) {
    return (uint32_t)(uintptr_t)p;
}

__device__ __forceinline__ void async_copy_b128(uint32_t lds_dst, const void* gsrc) {
    // CDNA5 async copy: copies 16 B from global memory to the LDS byte offset in VDST, tracked by ASYNCcnt.
    asm volatile("global_load_async_to_lds_b128 %0, %1, off"
                 :: "v"(lds_dst), "v"(gsrc)
                 : "memory");
}

__device__ __forceinline__ void wait_asynccnt0() {
#if __has_builtin(__builtin_amdgcn_s_wait_asynccnt)
    __builtin_amdgcn_s_wait_asynccnt(0);
#else
    asm volatile("s_wait_asynccnt 0x0" ::: "memory");
#endif
}

// ---- Kernel 1: split weight into bf16 hi/lo and compute ||w||^2 per row ----
__global__ __launch_bounds__(128)
void prep_w_kernel(const float* __restrict__ w,
                   uint16_t* __restrict__ w_hi,
                   uint16_t* __restrict__ w_lo,
                   float* __restrict__ w_sq) {
    __shared__ float red[128];
    const int k = blockIdx.x;
    const int t = threadIdx.x;
    float4 x = ((const float4*)(w + (size_t)k * D_MODEL))[t];

    uint16_t h0 = f32_to_bf16_rne(x.x);
    uint16_t h1 = f32_to_bf16_rne(x.y);
    uint16_t h2 = f32_to_bf16_rne(x.z);
    uint16_t h3 = f32_to_bf16_rne(x.w);

    uint2 ph, pl;
    ph.x = (uint32_t)h0 | ((uint32_t)h1 << 16);
    ph.y = (uint32_t)h2 | ((uint32_t)h3 << 16);
    pl.x = (uint32_t)f32_to_bf16_rne(x.x - bf16_bits_to_f32(h0)) |
           ((uint32_t)f32_to_bf16_rne(x.y - bf16_bits_to_f32(h1)) << 16);
    pl.y = (uint32_t)f32_to_bf16_rne(x.z - bf16_bits_to_f32(h2)) |
           ((uint32_t)f32_to_bf16_rne(x.w - bf16_bits_to_f32(h3)) << 16);
    ((uint2*)(w_hi + (size_t)k * D_MODEL))[t] = ph;
    ((uint2*)(w_lo + (size_t)k * D_MODEL))[t] = pl;

    red[t] = x.x * x.x + x.y * x.y + x.z * x.z + x.w * x.w;
    __syncthreads();
    for (int off = 64; off > 0; off >>= 1) {
        if (t < off) red[t] += red[t + off];
        __syncthreads();
    }
    if (t == 0) w_sq[k] = red[0];
}

__device__ __forceinline__ v8f wmma_bf16(v16bf a, v16bf b, v8f c) {
    return __builtin_amdgcn_wmma_f32_16x16x32_bf16(
        false, a, false, b, (short)0, c, false, false);
}

// Issue async fill of one double-buffer slot with B tiles for [nt, nt+1]:
// rows 0..31 = w_hi codes nt*16..nt*16+31, rows 32..63 = w_lo same codes.
__device__ __forceinline__ void fill_b_tiles(uint32_t lds_base,
                                             const uint16_t* __restrict__ w_hi,
                                             const uint16_t* __restrict__ w_lo,
                                             int nt, int tid) {
#pragma unroll
    for (int i = 0; i < 32; ++i) {
        const int c  = tid + i * THREADS_PER_WG;  // 0..4095 16B-chunks
        const int r  = c >> 6;                    // 0..63   row slot
        const int ch = c & 63;                    // chunk within row (8 halfs)
        const uint16_t* src = ((r < 32) ? w_hi : w_lo)
                            + (size_t)(nt * 16 + (r & 31)) * D_MODEL + ch * 8;
        async_copy_b128(lds_base + (uint32_t)(r * B_ROW_BYTES + ch * 16), src);
    }
}

__device__ __forceinline__ v16bf lds_load_b32B(const char* p) {
    v8bf lo = *(const v8bf*)(p);
    v8bf hi = *(const v8bf*)(p + 32);
    return __builtin_shufflevector(lo, hi,
        0,1,2,3,4,5,6,7,8,9,10,11,12,13,14,15);
}

// ---- Kernel 2: fused bf16x3 distance-argmin + gather ----
// waves_per_eu=2: cap VGPRs so 2 waves/SIMD stay resident (round-2 build used ~830).
__global__ __launch_bounds__(THREADS_PER_WG, 2)
void vq_argmin_kernel(const float* __restrict__ z,
                      const float* __restrict__ w,
                      const uint16_t* __restrict__ w_hi,
                      const uint16_t* __restrict__ w_lo,
                      const float* __restrict__ w_sq,
                      int* __restrict__ out_idx,
                      float* __restrict__ out_q) {
    extern __shared__ char smem[];                // 2 x B_BUF_BYTES double buffer

    const int tid    = threadIdx.x;
    const int lane   = tid & 31;
    const int waveId = tid >> 5;
    const int row0   = blockIdx.x * ROWS_PER_WG;
    const int ncol   = lane & 15;
    const int hiGrp  = (lane >= 16) ? 1 : 0;

    // ---- Kick off async fill of buffer 0 while we build A in registers ----
    const uint32_t smem_base = lds_offset(smem);
    fill_b_tiles(smem_base, w_hi, w_lo, /*nt=*/0, tid);

    // ---- A matrix (16x512) held entirely in VGPRs as bf16 hi/lo split ----
    // ISA A layout per K-slice of 32: lanes 0-15 row M=lane, halfs {K..K+7, K+16..K+23};
    // lanes 16-31 same rows, halfs {K+8..K+15, K+24..K+31}.
    const float* zrow = z + (size_t)(row0 + waveId * 16 + ncol) * D_MODEL;
    v16bf a_hi[16], a_lo[16];
#pragma unroll
    for (int ks = 0; ks < 16; ++ks) {
        const int cb = ks * 32 + hiGrp * 8;
        float f[16];
        *(float4*)(f + 0)  = *(const float4*)(zrow + cb);
        *(float4*)(f + 4)  = *(const float4*)(zrow + cb + 4);
        *(float4*)(f + 8)  = *(const float4*)(zrow + cb + 16);
        *(float4*)(f + 12) = *(const float4*)(zrow + cb + 20);
        v16us h, l;
#pragma unroll
        for (int i = 0; i < 16; ++i) {
            uint16_t hb = f32_to_bf16_rne(f[i]);
            h[i] = hb;
            l[i] = f32_to_bf16_rne(f[i] - bf16_bits_to_f32(hb));
        }
        a_hi[ks] = __builtin_bit_cast(v16bf, h);
        a_lo[ks] = __builtin_bit_cast(v16bf, l);
    }

    wait_asynccnt0();
    __syncthreads();

    float best[8];
    int   bidx[8];
#pragma unroll
    for (int r = 0; r < 8; ++r) { best[r] = 3.4e38f; bidx[r] = 0; }

    for (int nt = 0; nt < KCODES / 16; nt += 2) {
        const int cur = (nt >> 1) & 1;
        if (nt + 2 < KCODES / 16)     // prefetch next B tiles into other buffer
            fill_b_tiles(smem_base + (uint32_t)((cur ^ 1) * B_BUF_BYTES),
                         w_hi, w_lo, nt + 2, tid);

        const char* buf = smem + cur * B_BUF_BYTES;
        // B (32x16) ISA layout: lane n -> column n, K {0..7, 16..23};
        // lane 16+n -> K {8..15, 24..31}. Two 16B chunks per lane per K-slice.
        const char* b0h = buf + (ncol)      * B_ROW_BYTES + hiGrp * 16;
        const char* b1h = buf + (16 + ncol) * B_ROW_BYTES + hiGrp * 16;
        const char* b0l = buf + (32 + ncol) * B_ROW_BYTES + hiGrp * 16;
        const char* b1l = buf + (48 + ncol) * B_ROW_BYTES + hiGrp * 16;

        // 6 independent accumulator chains: the 6 WMMAs per K-step have no
        // mutual dependencies, so the matrix pipe never stalls on C operands.
        v8f acc0hh = {}, acc0hl = {}, acc0lh = {};
        v8f acc1hh = {}, acc1hl = {}, acc1lh = {};
#pragma unroll
        for (int ks = 0; ks < 16; ++ks) {
            const int bo = ks * 64;
            v16bf bh0 = lds_load_b32B(b0h + bo);
            v16bf bl0 = lds_load_b32B(b0l + bo);
            v16bf bh1 = lds_load_b32B(b1h + bo);
            v16bf bl1 = lds_load_b32B(b1l + bo);
            acc0hh = wmma_bf16(a_hi[ks], bh0, acc0hh);
            acc1hh = wmma_bf16(a_hi[ks], bh1, acc1hh);
            acc0hl = wmma_bf16(a_hi[ks], bl0, acc0hl);
            acc1hl = wmma_bf16(a_hi[ks], bl1, acc1hl);
            acc0lh = wmma_bf16(a_lo[ks], bh0, acc0lh);
            acc1lh = wmma_bf16(a_lo[ks], bh1, acc1lh);
        }

        const int c0 = nt * 16 + ncol;
        const int c1 = c0 + 16;
        const float q0 = w_sq[c0];
        const float q1 = w_sq[c1];
#pragma unroll
        for (int r = 0; r < 8; ++r) {
            float d0 = acc0hh[r] + acc0hl[r] + acc0lh[r];
            float s0 = q0 - 2.0f * d0;        // ||z||^2 constant per row: dropped
            if (s0 < best[r]) { best[r] = s0; bidx[r] = c0; }
            float d1 = acc1hh[r] + acc1hl[r] + acc1lh[r];
            float s1 = q1 - 2.0f * d1;
            if (s1 < best[r]) { best[r] = s1; bidx[r] = c1; }
        }

        wait_asynccnt0();     // our prefetch into the other buffer is complete
        __syncthreads();      // everyone done reading `cur` before it is refilled
    }

    // ---- Cross-lane argmin reduction (reuse LDS after the loop's barrier) ----
    float* sc  = (float*)smem;                               // [64][16]
    int*   id  = (int*)(smem + ROWS_PER_WG * 16 * 4);        // [64][16]
    int*   win = (int*)(smem + 2 * ROWS_PER_WG * 16 * 4);    // [64]
#pragma unroll
    for (int r = 0; r < 8; ++r) {
        const int rr = waveId * 16 + r + hiGrp * 8;  // C layout: M = r + 8*(lane>=16)
        sc[rr * 16 + ncol] = best[r];
        id[rr * 16 + ncol] = bidx[r];
    }
    __syncthreads();
    if (tid < ROWS_PER_WG) {
        float b  = sc[tid * 16];
        int   bi = id[tid * 16];
        for (int c = 1; c < 16; ++c) {
            float s  = sc[tid * 16 + c];
            int   i2 = id[tid * 16 + c];
            if (s < b || (s == b && i2 < bi)) { b = s; bi = i2; }  // first-index ties
        }
        win[tid] = bi;
        out_idx[row0 + tid] = bi;
    }
    __syncthreads();

    // ---- Gather quantized rows (exact fp32 copy of winning weight rows) ----
    for (int v = tid; v < ROWS_PER_WG * (D_MODEL / 4); v += THREADS_PER_WG) {
        const int r  = v >> 7;
        const int c4 = v & 127;
        const int k  = win[r];
        ((float4*)(out_q + (size_t)(row0 + r) * D_MODEL))[c4] =
            ((const float4*)(w + (size_t)k * D_MODEL))[c4];
    }
}

extern "C" void kernel_launch(void* const* d_in, const int* in_sizes, int n_in,
                              void* d_out, int out_size, void* d_ws, size_t ws_size,
                              hipStream_t stream) {
    const float* z = (const float*)d_in[0];
    const float* w = (const float*)d_in[1];
    const int N = in_sizes[0] / D_MODEL;   // 65536

    // Workspace layout: w_hi (4MB) | w_lo (4MB) | w_sq (16KB)
    uint16_t* w_hi = (uint16_t*)d_ws;
    uint16_t* w_lo = w_hi + (size_t)KCODES * D_MODEL;
    float*    w_sq = (float*)(w_lo + (size_t)KCODES * D_MODEL);

    // Output: [N] int32 indices, then [N*D] fp32 quantized, concatenated flat.
    int*   out_idx = (int*)d_out;
    float* out_q   = (float*)d_out + N;

    prep_w_kernel<<<KCODES, 128, 0, stream>>>(w, w_hi, w_lo, w_sq);

    const size_t shmem = (size_t)2 * B_BUF_BYTES;   // 133,120 B double buffer
    vq_argmin_kernel<<<N / ROWS_PER_WG, THREADS_PER_WG, shmem, stream>>>(
        z, w, w_hi, w_lo, w_sq, out_idx, out_q);
}
